// SpatialEncoder_7533372637869
// MI455X (gfx1250) — compile-verified, matrix operand order checked by reading, not run on hardware
//
#include <hip/hip_runtime.h>

typedef __attribute__((ext_vector_type(16))) __bf16 v16bf;
typedef __attribute__((ext_vector_type(8)))  float  v8f;

#define ENC_NEGINF 0x007FFFFFu  // order-preserving encoding of -inf

__device__ __forceinline__ unsigned fenc(float f) {
  unsigned u = __float_as_uint(f);
  return (u & 0x80000000u) ? ~u : (u | 0x80000000u);
}
__device__ __forceinline__ float fdec(unsigned u) {
  return (u & 0x80000000u) ? __uint_as_float(u ^ 0x80000000u)
                           : __uint_as_float(~u);
}
__device__ __forceinline__ float4 ld4(const float* __restrict__ p) {
  return *(const float4*)p;
}

// ---------------------------------------------------------------------------
// Dense GEMM: Y[n,M] = X[n,K] @ W[K,M] (+bias) (+=Y if accumulate)
// fp32 inputs split into bf16 hi/lo; 3x v_wmma_f32_16x16x32_bf16 per K-step
// (hi*hi + hi*lo + lo*hi) gives ~fp32 accuracy with fp32 accumulation.
// One wave per 16x16 tile; K multiple of 32 (all GEMMs here have K=128).
// ---------------------------------------------------------------------------
__global__ void gemm_wmma_bf16x3(const float* __restrict__ X,
                                 const float* __restrict__ W,
                                 const float* __restrict__ bias,
                                 float* __restrict__ Y,
                                 int n, int K, int M, int accumulate) {
  const int lane = threadIdx.x & 31;
  const int tiles_m = (n + 15) >> 4;
  const int tile = blockIdx.x;
  const int tm = tile % tiles_m;
  const int tn = tile / tiles_m;
  const int half = lane >> 4;   // 0: K-base 0, 1: K-base 8 (A/B bf16 layout)
  const int l16  = lane & 15;
  const int kb   = half * 8;
  int row  = tm * 16 + l16;
  int rowc = row < n ? row : n - 1;
  int col  = tn * 16 + l16;
  const float* Xr = X + (size_t)rowc * K;

  v8f acc = {};
  for (int k0 = 0; k0 < K; k0 += 32) {
    // A: two contiguous 8-float chunks per lane -> 4x float4
    float4 a0 = ld4(Xr + k0 + kb);
    float4 a1 = ld4(Xr + k0 + kb + 4);
    float4 a2 = ld4(Xr + k0 + kb + 16);
    float4 a3 = ld4(Xr + k0 + kb + 20);
    float af[16] = {a0.x, a0.y, a0.z, a0.w, a1.x, a1.y, a1.z, a1.w,
                    a2.x, a2.y, a2.z, a2.w, a3.x, a3.y, a3.z, a3.w};
    v16bf ahi, alo, bhi, blo;
#pragma unroll
    for (int i = 0; i < 16; ++i) {
      float av = af[i];
      __bf16 ah = (__bf16)av;
      ahi[i] = ah;
      alo[i] = (__bf16)(av - (float)ah);
      // B: column `col`, K-stride M (not vectorizable; cached in L0/L2)
      int koff = kb + (i & 7) + ((i >> 3) << 4);
      float bv = W[(size_t)(k0 + koff) * M + col];
      __bf16 bh = (__bf16)bv;
      bhi[i] = bh;
      blo[i] = (__bf16)(bv - (float)bh);
    }
    acc = __builtin_amdgcn_wmma_f32_16x16x32_bf16(false, ahi, false, bhi,
                                                  (short)0, acc, false, false);
    acc = __builtin_amdgcn_wmma_f32_16x16x32_bf16(false, ahi, false, blo,
                                                  (short)0, acc, false, false);
    acc = __builtin_amdgcn_wmma_f32_16x16x32_bf16(false, alo, false, bhi,
                                                  (short)0, acc, false, false);
  }
  float bv = bias ? bias[col] : 0.f;
  int gbase = tm * 16 + half * 8;
  if (accumulate) {
#pragma unroll
    for (int r = 0; r < 8; ++r) {
      int grow = gbase + r;
      if (grow < n) {
        size_t off = (size_t)grow * M + col;
        Y[off] = Y[off] + acc[r] + bv;
      }
    }
  } else {
#pragma unroll
    for (int r = 0; r < 8; ++r) {
      int grow = gbase + r;
      if (grow < n) Y[(size_t)grow * M + col] = acc[r] + bv;
    }
  }
}

// ---------------------------------------------------------------------------
__global__ void fill_f32x4(float4* __restrict__ p, float v, size_t cnt4) {
  size_t i = (size_t)blockIdx.x * blockDim.x + threadIdx.x;
  if (i < cnt4) p[i] = make_float4(v, v, v, v);
}
__global__ void fill_u32(unsigned* __restrict__ p, unsigned v, size_t cnt) {
  size_t i = (size_t)blockIdx.x * blockDim.x + threadIdx.x;
  if (i < cnt) p[i] = v;
}

// ---------------------------------------------------------------------------
// GATv2 edge logits (H==2, C==128): one wave per edge; each lane owns 4
// consecutive channels per head -> b128 loads, 512B/wave fully coalesced.
//   logit[e,h] = sum_c att[h,c]*leakyrelu(xl[src,h,c]+xr[tgt,h,c]+(ea@We)[h,c])
// ea@We computed on the fly (We is 16KB, cache-resident) -> saves 656MB/layer.
// ---------------------------------------------------------------------------
__global__ void edge_logits_k(const float* __restrict__ xl,
                              const float* __restrict__ xr,
                              const float* __restrict__ ea,
                              const float* __restrict__ We,
                              const float* __restrict__ att,
                              const int* __restrict__ src,
                              const int* __restrict__ tgt,
                              float* __restrict__ logits,
                              unsigned* __restrict__ nodemax,
                              int E, int C) {
  const int H = 2;
  int e = blockIdx.x * (blockDim.x >> 5) + (threadIdx.x >> 5);
  int lane = threadIdx.x & 31;
  if (e >= E) return;
  int s = src[e], t = tgt[e];
  const int HC = H * C;
  float4 ev[4];  // ED == 16 edge features
#pragma unroll
  for (int i = 0; i < 4; ++i) ev[i] = ld4(ea + (size_t)e * 16 + i * 4);
  const float eav[16] = {ev[0].x, ev[0].y, ev[0].z, ev[0].w,
                         ev[1].x, ev[1].y, ev[1].z, ev[1].w,
                         ev[2].x, ev[2].y, ev[2].z, ev[2].w,
                         ev[3].x, ev[3].y, ev[3].z, ev[3].w};
  float part[2];
#pragma unroll
  for (int h = 0; h < H; ++h) {
    int base = h * C + lane * 4;
    float4 a = ld4(xl + (size_t)s * HC + base);
    float4 b = ld4(xr + (size_t)t * HC + base);
    float vx = a.x + b.x, vy = a.y + b.y, vz = a.z + b.z, vw = a.w + b.w;
#pragma unroll
    for (int i = 0; i < 16; ++i) {
      float4 w = ld4(We + (size_t)i * HC + base);
      vx = fmaf(eav[i], w.x, vx);
      vy = fmaf(eav[i], w.y, vy);
      vz = fmaf(eav[i], w.z, vz);
      vw = fmaf(eav[i], w.w, vw);
    }
    vx = vx > 0.f ? vx : 0.2f * vx;
    vy = vy > 0.f ? vy : 0.2f * vy;
    vz = vz > 0.f ? vz : 0.2f * vz;
    vw = vw > 0.f ? vw : 0.2f * vw;
    float4 at = ld4(att + base);
    part[h] = vx * at.x + vy * at.y + vz * at.z + vw * at.w;
  }
#pragma unroll
  for (int off = 16; off >= 1; off >>= 1) {
    part[0] += __shfl_xor(part[0], off, 32);
    part[1] += __shfl_xor(part[1], off, 32);
  }
  if (lane == 0) {
    logits[(size_t)e * H + 0] = part[0];
    logits[(size_t)e * H + 1] = part[1];
    atomicMax(&nodemax[t * H + 0], fenc(part[0]));
    atomicMax(&nodemax[t * H + 1], fenc(part[1]));
  }
}

// exp(logit - segmax) and segment-sum via atomics; one thread per (e,h)
__global__ void edge_exp_k(const float* __restrict__ logits,
                           const int* __restrict__ tgt,
                           const unsigned* __restrict__ nodemax,
                           float* __restrict__ expb,
                           float* __restrict__ nodesum, int E, int H) {
  int i = blockIdx.x * blockDim.x + threadIdx.x;
  if (i >= E * H) return;
  int e = i / H, h = i - e * H;
  int t = tgt[e];
  float m = fdec(nodemax[t * H + h]);
  float v = expf(logits[i] - m);
  expb[i] = v;
  atomicAdd(&nodesum[t * H + h], v);
}

// accum[tgt,h,c] += xl[src,h,c] * alpha[e,h]; one wave per edge, b128 gathers
__global__ void edge_scatter_k(const float* __restrict__ xl,
                               const float* __restrict__ expb,
                               const float* __restrict__ nodesum,
                               const int* __restrict__ src,
                               const int* __restrict__ tgt,
                               float* __restrict__ accum, int E, int C) {
  const int H = 2;
  int e = blockIdx.x * (blockDim.x >> 5) + (threadIdx.x >> 5);
  int lane = threadIdx.x & 31;
  if (e >= E) return;
  int s = src[e], t = tgt[e];
  const int HC = H * C;
#pragma unroll
  for (int h = 0; h < H; ++h) {
    float coef = expb[(size_t)e * H + h] / (nodesum[t * H + h] + 1e-16f);
    int base = h * C + lane * 4;
    float4 xv = ld4(xl + (size_t)s * HC + base);
    float* ap = accum + (size_t)t * HC + base;
    atomicAdd(ap + 0, xv.x * coef);
    atomicAdd(ap + 1, xv.y * coef);
    atomicAdd(ap + 2, xv.z * coef);
    atomicAdd(ap + 3, xv.w * coef);
  }
}

// ---------------------------------------------------------------------------
// out = gelu(LN(mean_h(acc_in) + bias + res)); one wave per node, C==128,
// each lane owns 4 consecutive channels. H==1 + bias=null serves SAGE tail.
// ---------------------------------------------------------------------------
__global__ void node_norm_gelu_k(const float* __restrict__ acc_in, int H,
                                 const float* __restrict__ bias,
                                 const float* __restrict__ res,
                                 const float* __restrict__ g,
                                 const float* __restrict__ b,
                                 float* __restrict__ out, int n, int C) {
  int nid = blockIdx.x * (blockDim.x >> 5) + (threadIdx.x >> 5);
  int lane = threadIdx.x & 31;
  if (nid >= n) return;
  int base = lane * 4;
  float invH = 1.f / (float)H;
  float4 v = make_float4(0.f, 0.f, 0.f, 0.f);
  for (int h = 0; h < H; ++h) {
    float4 a = ld4(acc_in + ((size_t)nid * H + h) * C + base);
    v.x += a.x; v.y += a.y; v.z += a.z; v.w += a.w;
  }
  v.x *= invH; v.y *= invH; v.z *= invH; v.w *= invH;
  if (bias) {
    float4 bb = ld4(bias + base);
    v.x += bb.x; v.y += bb.y; v.z += bb.z; v.w += bb.w;
  }
  if (res) {
    float4 r = ld4(res + (size_t)nid * C + base);
    v.x += r.x; v.y += r.y; v.z += r.z; v.w += r.w;
  }
  float msum = v.x + v.y + v.z + v.w;
#pragma unroll
  for (int off = 16; off >= 1; off >>= 1) msum += __shfl_xor(msum, off, 32);
  float mean = msum / (float)C;
  float dx = v.x - mean, dy = v.y - mean, dz = v.z - mean, dw = v.w - mean;
  float vs = dx * dx + dy * dy + dz * dz + dw * dw;
#pragma unroll
  for (int off = 16; off >= 1; off >>= 1) vs += __shfl_xor(vs, off, 32);
  float rs = rsqrtf(vs / (float)C + 1e-5f);
  float4 gg = ld4(g + base), bb2 = ld4(b + base);
  float t0 = dx * rs * gg.x + bb2.x;
  float t1 = dy * rs * gg.y + bb2.y;
  float t2 = dz * rs * gg.z + bb2.z;
  float t3 = dw * rs * gg.w + bb2.w;
  const float is2 = 0.70710678118f;
  float4 o;
  o.x = 0.5f * t0 * (1.f + erff(t0 * is2));
  o.y = 0.5f * t1 * (1.f + erff(t1 * is2));
  o.z = 0.5f * t2 * (1.f + erff(t2 * is2));
  o.w = 0.5f * t3 * (1.f + erff(t3 * is2));
  *(float4*)(out + (size_t)nid * C + base) = o;
}

// SAGE max aggregation: atomic segment-max of h2[src] at tgt (uint-encoded)
__global__ void sage_max_k(const float* __restrict__ h2,
                           const int* __restrict__ src,
                           const int* __restrict__ tgt,
                           unsigned* __restrict__ nenc, int E, int C) {
  int e = blockIdx.x * (blockDim.x >> 5) + (threadIdx.x >> 5);
  int lane = threadIdx.x & 31;
  if (e >= E) return;
  int s = src[e], t = tgt[e];
  int base = lane * 4;  // C == 128
  float4 hv = ld4(h2 + (size_t)s * C + base);
  unsigned* np = nenc + (size_t)t * C + base;
  atomicMax(np + 0, fenc(hv.x));
  atomicMax(np + 1, fenc(hv.y));
  atomicMax(np + 2, fenc(hv.z));
  atomicMax(np + 3, fenc(hv.w));
}
__global__ void sage_decode_k(const unsigned* __restrict__ nenc,
                              float* __restrict__ neigh, size_t cnt) {
  size_t i = (size_t)blockIdx.x * blockDim.x + threadIdx.x;
  if (i < cnt) {
    unsigned u = nenc[i];
    neigh[i] = (u == ENC_NEGINF) ? 0.f : fdec(u);  // empty segment -> 0
  }
}

// ---------------------------------------------------------------------------
extern "C" void kernel_launch(void* const* d_in, const int* in_sizes, int n_in,
                              void* d_out, int out_size, void* d_ws, size_t ws_size,
                              hipStream_t stream) {
  const int N = 20000, E = 320000, C = 128, H = 2, IN = 128;
  (void)in_sizes; (void)n_in; (void)out_size; (void)ws_size;

  const float* x     = (const float*)d_in[0];
  const int*   ei    = (const int*)  d_in[1];
  const float* ea    = (const float*)d_in[2];
  const float* Wl1   = (const float*)d_in[3];  const float* bl1 = (const float*)d_in[4];
  const float* Wr1   = (const float*)d_in[5];  const float* br1 = (const float*)d_in[6];
  const float* We1   = (const float*)d_in[7];  const float* att1= (const float*)d_in[8];
  const float* bias1 = (const float*)d_in[9];  const float* Wres1=(const float*)d_in[10];
  const float* ln1g  = (const float*)d_in[11]; const float* ln1b= (const float*)d_in[12];
  const float* Wl2   = (const float*)d_in[13]; const float* bl2 = (const float*)d_in[14];
  const float* Wr2   = (const float*)d_in[15]; const float* br2 = (const float*)d_in[16];
  const float* We2   = (const float*)d_in[17]; const float* att2= (const float*)d_in[18];
  const float* bias2 = (const float*)d_in[19]; const float* Wres2=(const float*)d_in[20];
  const float* ln2g  = (const float*)d_in[21]; const float* ln2b= (const float*)d_in[22];
  const float* Wsl   = (const float*)d_in[23]; const float* bsl = (const float*)d_in[24];
  const float* Wsr   = (const float*)d_in[25];
  const float* ln3g  = (const float*)d_in[26]; const float* ln3b= (const float*)d_in[27];

  const int* src = ei;        // edge_index row 0
  const int* tgt = ei + E;    // edge_index row 1

  // bump allocator on workspace
  char* wsb = (char*)d_ws;
  size_t off = 0;
  auto alloc = [&](size_t bytes) -> char* {
    char* p = wsb + off;
    off += (bytes + 255) & ~(size_t)255;
    return p;
  };
  const size_t HC = (size_t)H * C;
  float*    xl      = (float*)   alloc((size_t)N * HC * 4);
  float*    xr      = (float*)   alloc((size_t)N * HC * 4);
  float*    accum   = (float*)   alloc((size_t)N * HC * 4);
  float*    logits  = (float*)   alloc((size_t)E * H * 4);
  float*    expb    = (float*)   alloc((size_t)E * H * 4);
  unsigned* nodemax = (unsigned*)alloc((size_t)N * H * 4);
  float*    nodesum = (float*)   alloc((size_t)N * H * 4);
  float*    res     = (float*)   alloc((size_t)N * C * 4);
  float*    h1      = (float*)   alloc((size_t)N * C * 4);
  float*    h2      = (float*)   alloc((size_t)N * C * 4);
  // aliases (gat buffers are dead by the SAGE stage)
  unsigned* nenc  = (unsigned*)accum;
  float*    neigh = xl;
  float*    Ybuf  = xr;

  const int tilesA = ((N + 15) / 16) * ((H * C) / 16);  // M = 256
  const int tilesB = ((N + 15) / 16) * (C / 16);        // M = 128
  const int edgeBlocks = (E + 7) / 8;
  const int nodeBlocks = (N + 7) / 8;
  const int nhc4Blocks = (int)(((size_t)N * HC / 4 + 255) / 256);
  const int nh4Blocks  = (int)(((size_t)N * H / 4 + 255) / 256);
  const int ncBlocks   = (int)(((size_t)N * C + 255) / 256);
  const int nhBlocks   = (N * H + 255) / 256;
  const int ehBlocks   = (E * H + 255) / 256;

  // ---- GAT layer 1 ----
  gemm_wmma_bf16x3<<<tilesA, 32, 0, stream>>>(x, Wl1, bl1, xl, N, IN, H * C, 0);
  gemm_wmma_bf16x3<<<tilesA, 32, 0, stream>>>(x, Wr1, br1, xr, N, IN, H * C, 0);
  gemm_wmma_bf16x3<<<tilesB, 32, 0, stream>>>(x, Wres1, nullptr, res, N, IN, C, 0);
  fill_u32<<<nhBlocks, 256, 0, stream>>>(nodemax, ENC_NEGINF, (size_t)N * H);
  fill_f32x4<<<nh4Blocks, 256, 0, stream>>>((float4*)nodesum, 0.f, (size_t)N * H / 4);
  fill_f32x4<<<nhc4Blocks, 256, 0, stream>>>((float4*)accum, 0.f, (size_t)N * HC / 4);
  edge_logits_k<<<edgeBlocks, 256, 0, stream>>>(xl, xr, ea, We1, att1, src, tgt,
                                                logits, nodemax, E, C);
  edge_exp_k<<<ehBlocks, 256, 0, stream>>>(logits, tgt, nodemax, expb, nodesum, E, H);
  edge_scatter_k<<<edgeBlocks, 256, 0, stream>>>(xl, expb, nodesum, src, tgt, accum, E, C);
  node_norm_gelu_k<<<nodeBlocks, 256, 0, stream>>>(accum, H, bias1, res, ln1g, ln1b,
                                                   h1, N, C);

  // ---- GAT layer 2 ----
  gemm_wmma_bf16x3<<<tilesA, 32, 0, stream>>>(h1, Wl2, bl2, xl, N, C, H * C, 0);
  gemm_wmma_bf16x3<<<tilesA, 32, 0, stream>>>(h1, Wr2, br2, xr, N, C, H * C, 0);
  gemm_wmma_bf16x3<<<tilesB, 32, 0, stream>>>(h1, Wres2, nullptr, res, N, C, C, 0);
  fill_u32<<<nhBlocks, 256, 0, stream>>>(nodemax, ENC_NEGINF, (size_t)N * H);
  fill_f32x4<<<nh4Blocks, 256, 0, stream>>>((float4*)nodesum, 0.f, (size_t)N * H / 4);
  fill_f32x4<<<nhc4Blocks, 256, 0, stream>>>((float4*)accum, 0.f, (size_t)N * HC / 4);
  edge_logits_k<<<edgeBlocks, 256, 0, stream>>>(xl, xr, ea, We2, att2, src, tgt,
                                                logits, nodemax, E, C);
  edge_exp_k<<<ehBlocks, 256, 0, stream>>>(logits, tgt, nodemax, expb, nodesum, E, H);
  edge_scatter_k<<<edgeBlocks, 256, 0, stream>>>(xl, expb, nodesum, src, tgt, accum, E, C);
  node_norm_gelu_k<<<nodeBlocks, 256, 0, stream>>>(accum, H, bias2, res, ln2g, ln2b,
                                                   h2, N, C);

  // ---- SAGE (max aggregation) ----
  fill_u32<<<ncBlocks, 256, 0, stream>>>(nenc, ENC_NEGINF, (size_t)N * C);
  sage_max_k<<<edgeBlocks, 256, 0, stream>>>(h2, src, tgt, nenc, E, C);
  sage_decode_k<<<ncBlocks, 256, 0, stream>>>(nenc, neigh, (size_t)N * C);
  gemm_wmma_bf16x3<<<tilesB, 32, 0, stream>>>(neigh, Wsl, bsl, Ybuf, N, C, C, 0);
  gemm_wmma_bf16x3<<<tilesB, 32, 0, stream>>>(h2, Wsr, nullptr, Ybuf, N, C, C, 1);
  node_norm_gelu_k<<<nodeBlocks, 256, 0, stream>>>(Ybuf, 1, nullptr, h2, ln3g, ln3b,
                                                   (float*)d_out, N, C);
}
